// TNet_48326972015345
// MI455X (gfx1250) — matrix operand order repaired and checked by reading, not compiled, not run on hardware
//
#include <hip/hip_runtime.h>
#include <hip/hip_bf16.h>

// ---------------------------------------------------------------------------
// TNet forward for MI455X (gfx1250, wave32, WMMA + async global->LDS DMA).
//   h   = relu(relu(x) @ W_fc^T + b_fc)                 [8192, 512]
//   d   = ||h||^2 - 2 h@W_q^T + ||w_j||^2, clamp >= 0   [8192, 1024]
//   t   = -log1p(d)   (NU == 1)
//   out = t - logsumexp(t, axis=-1)
// Bulk GEMMs in f16 WMMA (fp32 accum), everything else fp32.
// ---------------------------------------------------------------------------

typedef _Float16 v16h  __attribute__((ext_vector_type(16)));
typedef float    v8f   __attribute__((ext_vector_type(8)));
typedef _Float16 f16x8 __attribute__((ext_vector_type(8)));
typedef _Float16 f16x4 __attribute__((ext_vector_type(4)));
typedef float    f32x4 __attribute__((ext_vector_type(4)));
typedef int      v4i   __attribute__((ext_vector_type(4)));

#define TN_B  8192
#define TN_NX 2048
#define TN_NH 512
#define TN_NY 1024

// ------------------- gfx1250 async global->LDS primitives ------------------
// Builtin signature (probed via diagnostics): (AS1 v4i*, AS3 v4i*, imm, imm).
// LDS operand wants the raw 32-bit LDS offset; for a generic __shared__
// pointer the low 32 bits ARE that offset, so we
// synthesize address-space pointers through integer casts.

__device__ __forceinline__ void async_cp16(const void* g, void* l) {
#if __has_builtin(__builtin_amdgcn_global_load_async_to_lds_b128)
  __builtin_amdgcn_global_load_async_to_lds_b128(
      (__attribute__((address_space(1))) v4i*)(uintptr_t)g,
      (__attribute__((address_space(3))) v4i*)(uint32_t)(uintptr_t)l,
      0, 0);
#else
  asm volatile("global_load_async_to_lds_b128 %0, %1, off"
               :: "v"((uint32_t)(uintptr_t)l), "v"(g)
               : "memory");
#endif
}

__device__ __forceinline__ void wait_async4() {
#if __has_builtin(__builtin_amdgcn_s_wait_asynccnt)
  __builtin_amdgcn_s_wait_asynccnt(4);
#else
  asm volatile("s_wait_asynccnt 4" ::: "memory");
#endif
}
__device__ __forceinline__ void wait_async0() {
#if __has_builtin(__builtin_amdgcn_s_wait_asynccnt)
  __builtin_amdgcn_s_wait_asynccnt(0);
#else
  asm volatile("s_wait_asynccnt 0" ::: "memory");
#endif
}

// ----------------------------- convert kernels -----------------------------

template<bool RELU>
__global__ __launch_bounds__(256) void k_cvt(const float* __restrict__ src,
                                             _Float16* __restrict__ dst,
                                             int n4) {
  int i = blockIdx.x * 256 + threadIdx.x;
  if (i >= n4) return;
  f32x4 v = ((const f32x4*)src)[i];
  f16x4 o;
#pragma unroll
  for (int j = 0; j < 4; ++j) {
    float f = v[j];
    if (RELU) f = fmaxf(f, 0.0f);
    o[j] = (_Float16)f;
  }
  ((f16x4*)dst)[i] = o;
}

// W_q: convert row to f16 and compute ||w_j||^2 (fp32) per row of 512.
__global__ __launch_bounds__(128) void k_wq(const float* __restrict__ Wq,
                                            _Float16* __restrict__ wq16,
                                            float* __restrict__ wq2) {
  __shared__ float red[128];
  const int row = blockIdx.x;
  const float* src = Wq + (long)row * TN_NH;
  _Float16* dst = wq16 + (long)row * TN_NH;
  float s = 0.f;
  for (int j = threadIdx.x; j < TN_NH; j += 128) {
    float v = src[j];
    s += v * v;
    dst[j] = (_Float16)v;
  }
  red[threadIdx.x] = s;
  __syncthreads();
  for (int off = 64; off > 0; off >>= 1) {
    if (threadIdx.x < off) red[threadIdx.x] += red[threadIdx.x + off];
    __syncthreads();
  }
  if (threadIdx.x == 0) wq2[row] = red[0];
}

// ------------------------------- WMMA GEMM ---------------------------------
// C[M,N] = A[M,K] @ W[N,K]^T, f16 inputs, fp32 accumulation.
// Block tile 128(M) x 128(N), K step 32, double-buffered LDS filled with
// GLOBAL_LOAD_ASYNC_TO_LDS_B128. 256 threads = 8 waves in a 4(M) x 2(N) grid;
// each wave owns a 32x64 patch = 2x4 v_wmma_f32_16x16x32_f16 tiles.
// MODE 0: out_h = relu(acc + bias[n])  (f16)
// MODE 1: out_f = -log1p(max(hh[m] - 2*acc + wq2[n], 0))  (f32)
template<int K, int N, int MODE>
__global__ __launch_bounds__(256) void k_gemm(const _Float16* __restrict__ A,
                                              const _Float16* __restrict__ W,
                                              const float* __restrict__ e0,
                                              const float* __restrict__ e1,
                                              _Float16* __restrict__ outh,
                                              float* __restrict__ outf) {
  constexpr int BM = 128, BN = 128, BK = 32;
  __shared__ _Float16 sA[2][BM * BK];  // [m][k], 8 KB per buffer
  __shared__ _Float16 sB[2][BN * BK];  // [n][k], 8 KB per buffer

  const int tid  = threadIdx.x;
  const int lane = tid & 31;
  const int wave = tid >> 5;
  const int wm   = wave & 3;    // 0..3 -> M direction (x32)
  const int wn   = wave >> 2;   // 0..1 -> N direction (x64)
  const int l    = lane & 15;
  const int hi   = lane >> 4;

  const long rowA0 = (long)blockIdx.x * BM;
  const long colB0 = (long)blockIdx.y * BN;

  // Staging: A tile = 512 16B chunks, B tile = 512 16B chunks; each thread
  // owns chunks {tid, tid+256} of each. chunk -> (row = c>>2, kk = (c&3)*8).
  const int r0 = tid >> 2;            // 0..63
  const int kk = (tid & 3) * 8;       // 0/8/16/24
  const int r1 = r0 + 64;
  const int d0 = r0 * BK + kk;        // LDS half-offset, 16B aligned
  const int d1 = r1 * BK + kk;

  const _Float16* gA0 = A + (rowA0 + r0) * (long)K + kk;
  const _Float16* gA1 = A + (rowA0 + r1) * (long)K + kk;
  const _Float16* gB0 = W + (colB0 + r0) * (long)K + kk;
  const _Float16* gB1 = W + (colB0 + r1) * (long)K + kk;

  v8f c[2][4] = {};

  constexpr int NSTEP = K / BK;

  // prologue: stage 0 into buffer 0
  async_cp16(gA0, &sA[0][d0]);
  async_cp16(gA1, &sA[0][d1]);
  async_cp16(gB0, &sB[0][d0]);
  async_cp16(gB1, &sB[0][d1]);

  for (int s = 0; s < NSTEP; ++s) {
    const int cur = s & 1;
    gA0 += BK; gA1 += BK; gB0 += BK; gB1 += BK;
    if (s + 1 < NSTEP) {
      // Overwrite of buffer 1-cur is safe: end-of-step barrier of step s-1
      // guaranteed all waves finished reading it.
      async_cp16(gA0, &sA[1 - cur][d0]);
      async_cp16(gA1, &sA[1 - cur][d1]);
      async_cp16(gB0, &sB[1 - cur][d0]);
      async_cp16(gB1, &sB[1 - cur][d1]);
      wait_async4();   // drain stage-s ops (in-order), keep 4 next-stage in flight
    } else {
      wait_async0();
    }
    __syncthreads();   // all waves' stage-s tiles resident

    // ---- build WMMA fragments from LDS (contiguous k = 16*hi + s, same in A and B) ----
    v16h a[2], b[4];
#pragma unroll
    for (int ti = 0; ti < 2; ++ti) {
      const _Float16* pa = &sA[cur][(wm * 32 + ti * 16 + l) * BK];
      // lanes 0-15: K 0..15 ; lanes 16-31: K 16..31 (contiguous)
      f16x8 alo = *(const f16x8*)(pa + hi * 16);
      f16x8 ahi = *(const f16x8*)(pa + hi * 16 + 8);
#pragma unroll
      for (int j = 0; j < 8; ++j) { a[ti][j] = alo[j]; a[ti][8 + j] = ahi[j]; }
    }
#pragma unroll
    for (int tj = 0; tj < 4; ++tj) {
      // lane group hi holds K = hi*16 .. hi*16+15 contiguously, N = l
      const _Float16* pb = &sB[cur][(wn * 64 + tj * 16 + l) * BK + hi * 16];
      b[tj] = *(const v16h*)pb;
    }

#pragma unroll
    for (int ti = 0; ti < 2; ++ti)
#pragma unroll
      for (int tj = 0; tj < 4; ++tj)
        c[ti][tj] = __builtin_amdgcn_wmma_f32_16x16x32_f16(
            false, a[ti], false, b[tj], (short)0, c[ti][tj], false, false);

    __syncthreads();   // all waves done reading buffer cur
  }

  // ------------------------------ epilogue ---------------------------------
#pragma unroll
  for (int ti = 0; ti < 2; ++ti) {
#pragma unroll
    for (int tj = 0; tj < 4; ++tj) {
      const long gn = colB0 + wn * 64 + tj * 16 + l;
#pragma unroll
      for (int v = 0; v < 8; ++v) {
        const long gm = rowA0 + wm * 32 + ti * 16 + v + hi * 8;
        const float acc = c[ti][tj][v];
        if (MODE == 0) {
          float val = fmaxf(acc + e0[gn], 0.0f);
          outh[gm * (long)N + gn] = (_Float16)val;
        } else {
          float d = e0[gm] - 2.0f * acc + e1[gn];
          d = fmaxf(d, 0.0f);
          outf[gm * (long)N + gn] = -log1pf(d);
        }
      }
    }
  }
}

// --------------------------- row norms of h --------------------------------
__global__ __launch_bounds__(256) void k_hh(const _Float16* __restrict__ h,
                                            float* __restrict__ hh) {
  const int row  = blockIdx.x * 8 + (threadIdx.x >> 5);
  const int lane = threadIdx.x & 31;
  const f16x8* p = (const f16x8*)(h + (long)row * TN_NH);
  float s = 0.f;
#pragma unroll
  for (int c = 0; c < 2; ++c) {
    f16x8 v = p[lane + 32 * c];
#pragma unroll
    for (int j = 0; j < 8; ++j) { float f = (float)v[j]; s += f * f; }
  }
#pragma unroll
  for (int m = 16; m > 0; m >>= 1) s += __shfl_xor(s, m, 32);
  if (lane == 0) hh[row] = s;
}

// ------------------------ row log-softmax (in place) -----------------------
__global__ __launch_bounds__(256) void k_lsm(float* __restrict__ t) {
  const int lane = threadIdx.x & 31;
  const int wave = threadIdx.x >> 5;
  float* p = t + (long)blockIdx.x * TN_NY;

  float v[4];
  float mx = -3.402823466e+38f;
#pragma unroll
  for (int j = 0; j < 4; ++j) {
    v[j] = p[threadIdx.x + 256 * j];
    mx = fmaxf(mx, v[j]);
  }
#pragma unroll
  for (int m = 16; m > 0; m >>= 1) mx = fmaxf(mx, __shfl_xor(mx, m, 32));

  __shared__ float smax[8];
  __shared__ float ssum[8];
  if (lane == 0) smax[wave] = mx;
  __syncthreads();
  float bm = smax[0];
#pragma unroll
  for (int i = 1; i < 8; ++i) bm = fmaxf(bm, smax[i]);

  float s = 0.f;
#pragma unroll
  for (int j = 0; j < 4; ++j) s += __expf(v[j] - bm);
#pragma unroll
  for (int m = 16; m > 0; m >>= 1) s += __shfl_xor(s, m, 32);
  if (lane == 0) ssum[wave] = s;
  __syncthreads();
  float bs = 0.f;
#pragma unroll
  for (int i = 0; i < 8; ++i) bs += ssum[i];

  const float lse = bm + __logf(bs);
#pragma unroll
  for (int j = 0; j < 4; ++j) p[threadIdx.x + 256 * j] = v[j] - lse;
}

// ------------------------------- launcher ----------------------------------
extern "C" void kernel_launch(void* const* d_in, const int* in_sizes, int n_in,
                              void* d_out, int out_size, void* d_ws, size_t ws_size,
                              hipStream_t stream) {
  (void)in_sizes; (void)n_in; (void)out_size; (void)ws_size;

  const float* x   = (const float*)d_in[0];  // [8192, 2048]
  const float* Wfc = (const float*)d_in[1];  // [512, 2048]
  const float* bfc = (const float*)d_in[2];  // [512]
  const float* Wq  = (const float*)d_in[3];  // [1024, 512]
  float* out = (float*)d_out;                // [8192, 1024]

  char* ws = (char*)d_ws;
  _Float16* xr    = (_Float16*)ws; ws += (size_t)TN_B  * TN_NX * 2;  // 32 MB
  _Float16* h     = (_Float16*)ws; ws += (size_t)TN_B  * TN_NH * 2;  //  8 MB
  _Float16* wfc16 = (_Float16*)ws; ws += (size_t)TN_NH * TN_NX * 2;  //  2 MB
  _Float16* wq16  = (_Float16*)ws; ws += (size_t)TN_NY * TN_NH * 2;  //  1 MB
  float*    hh    = (float*)ws;    ws += (size_t)TN_B * 4;
  float*    wq2   = (float*)ws;    ws += (size_t)TN_NY * 4;

  // 1) precision conversions + class-prototype norms
  {
    int n4 = TN_B * TN_NX / 4;
    k_cvt<true><<<(n4 + 255) / 256, 256, 0, stream>>>(x, xr, n4);
  }
  {
    int n4 = TN_NH * TN_NX / 4;
    k_cvt<false><<<(n4 + 255) / 256, 256, 0, stream>>>(Wfc, wfc16, n4);
  }
  k_wq<<<TN_NY, 128, 0, stream>>>(Wq, wq16, wq2);

  // 2) GEMM1: h = relu(xr @ Wfc^T + b)
  k_gemm<TN_NX, TN_NH, 0><<<dim3(TN_B / 128, TN_NH / 128), 256, 0, stream>>>(
      xr, wfc16, bfc, nullptr, h, nullptr);

  // 3) hh[m] = ||h_m||^2
  k_hh<<<TN_B / 8, 256, 0, stream>>>(h, hh);

  // 4) GEMM2 + distance + -log1p  -> d_out
  k_gemm<TN_NH, TN_NY, 1><<<dim3(TN_B / 128, TN_NY / 128), 256, 0, stream>>>(
      h, wq16, hh, wq2, nullptr, out);

  // 5) row-wise log-softmax normalization in place
  k_lsm<<<TN_B, 256, 0, stream>>>(out);
}
